// Fusion_35287451304478
// MI455X (gfx1250) — matrix operand order repaired and checked by reading, not compiled, not run on hardware
//
#include <hip/hip_runtime.h>

typedef _Float16 v16h __attribute__((ext_vector_type(16)));
typedef float    v8f  __attribute__((ext_vector_type(8)));
typedef int      vs4i __attribute__((vector_size(16)));   // async builtin's pointee type

#define AS1 __attribute__((address_space(1)))
#define AS3 __attribute__((address_space(3)))

#define S_LEN  2048
#define DHEAD  64
#define NHEADS 16
#define QTILE  128   // 8 waves * 16 rows
#define KCHUNK 64
#define NCHUNK (S_LEN / KCHUNK)
#define LDSTR  72    // padded LDS row stride (halves): 36 dwords -> conflict-free
#define LOG2E  1.4426950408889634f

#ifndef USE_ASYNC_LDS
#if __has_builtin(__builtin_amdgcn_global_load_async_to_lds_b128)
#define USE_ASYNC_LDS 1
#else
#define USE_ASYNC_LDS 0
#endif
#endif

union H16 { v16h v; _Float16 h[16]; unsigned u[8]; };
union PK2 { _Float16 h[2]; unsigned u; };

static __device__ inline v8f wmma_f16(v16h a, v16h b, v8f c) {
  return __builtin_amdgcn_wmma_f32_16x16x32_f16(
      false, a, false, b, (short)0, c, false, false);
}

// DPP16 ROW_XMASK butterfly step (stays within each 16-lane row)
template <int M>
static __device__ inline float bfly_max(float v) {
  int t = __builtin_amdgcn_update_dpp(0, __float_as_int(v), 0x160 + M, 0xf, 0xf, true);
  return fmaxf(v, __int_as_float(t));
}
template <int M>
static __device__ inline float bfly_add(float v) {
  int t = __builtin_amdgcn_update_dpp(0, __float_as_int(v), 0x160 + M, 0xf, 0xf, true);
  return v + __int_as_float(t);
}

// ---------------- prepass 1: K f32 -> f16 (row-major copy) ----------------
__global__ __launch_bounds__(256)
void cvt_k_kernel(const float* __restrict__ K, _Float16* __restrict__ Kh, int n4) {
  const int stride = gridDim.x * blockDim.x;
  for (int i = blockIdx.x * blockDim.x + threadIdx.x; i < n4; i += stride) {
    float4 f = ((const float4*)K)[i];
    PK2 a, b;
    a.h[0] = (_Float16)f.x; a.h[1] = (_Float16)f.y;
    b.h[0] = (_Float16)f.z; b.h[1] = (_Float16)f.w;
    uint2 o; o.x = a.u; o.y = b.u;
    ((uint2*)Kh)[i] = o;
  }
}

// -------- prepass 2: V f32 [bh][s][d] -> f16 transposed [bh][d][s] --------
__global__ __launch_bounds__(256)
void cvt_vt_kernel(const float* __restrict__ V, _Float16* __restrict__ Vth) {
  __shared__ _Float16 Tt[DHEAD][LDSTR];
  const int tid = threadIdx.x;
  const int s0  = blockIdx.x * 64;
  const int bh  = blockIdx.y;
  const size_t base = (size_t)bh * S_LEN * DHEAD;
  const int row0 = (tid * 4) / DHEAD, col = (tid * 4) % DHEAD;
#pragma unroll
  for (int p4 = 0; p4 < 4; ++p4) {
    const int r = row0 + p4 * 16;
    float4 vf = *(const float4*)(V + base + (size_t)(s0 + r) * DHEAD + col);
    Tt[col][r]   = (_Float16)vf.x;  Tt[col+1][r] = (_Float16)vf.y;
    Tt[col+2][r] = (_Float16)vf.z;  Tt[col+3][r] = (_Float16)vf.w;
  }
  __syncthreads();
#pragma unroll
  for (int i = 0; i < 2; ++i) {
    const int id = tid * 2 + i;          // 0..511
    const int d = id >> 3, seg = id & 7; // 64 rows x 8 16B segments
    uint4 val = *(const uint4*)&Tt[d][seg * 8];
    *(uint4*)(Vth + ((size_t)bh * DHEAD + d) * S_LEN + s0 + seg * 8) = val;
  }
}

// ------------------------------ main kernel -------------------------------
__global__ __launch_bounds__(256)
void fa_fwd_kernel(const float* __restrict__ Q, const _Float16* __restrict__ Kh,
                   const _Float16* __restrict__ Vth, const float* __restrict__ M,
                   float* __restrict__ Out)
{
  __shared__ _Float16 Kt[2][KCHUNK][LDSTR];  // ping-pong, key-major
  __shared__ _Float16 Vt[2][DHEAD][LDSTR];   // ping-pong, d-major (transposed)
  __shared__ _Float16 Pb[8][16][KCHUNK];     // per-wave P staging (C->A relayout)

  const int tid  = threadIdx.x;
  const int lane = tid & 31;
  const int wave = tid >> 5;
  const int bh   = blockIdx.y;
  const int bidx = bh / NHEADS;

  const int qbase = blockIdx.x * QTILE + wave * 16;
  const int n     = lane & 15;
  const int hi    = lane >> 4;
  const int a_b0  = hi * 8;          // A-layout K offset / C-layout row offset
  const int b_b0  = hi * 16;         // unused: B feeds use a_b0, same k order as A

  const size_t base = (size_t)bh * S_LEN * DHEAD;

  // issue the async global->LDS copies for chunk `it` into buffer it&1
  auto issueStage = [&](int it) {
    const int kc0 = it * KCHUNK, buf = it & 1;
#pragma unroll
    for (int i = 0; i < 2; ++i) {
      const int id  = tid * 2 + i;        // 0..511: 64 rows x 8 16B segments
      const int row = id >> 3, seg = id & 7;
      const _Float16* gk = Kh + base + (size_t)(kc0 + row) * DHEAD + seg * 8;
      const _Float16* gv = Vth + base + (size_t)row * S_LEN + kc0 + seg * 8;
#if USE_ASYNC_LDS
      __builtin_amdgcn_global_load_async_to_lds_b128(
          (AS1 vs4i*)(void*)gk, (AS3 vs4i*)(void*)&Kt[buf][row][seg * 8], 0, 0);
      __builtin_amdgcn_global_load_async_to_lds_b128(
          (AS1 vs4i*)(void*)gv, (AS3 vs4i*)(void*)&Vt[buf][row][seg * 8], 0, 0);
#else
      *(uint4*)&Kt[buf][row][seg * 8] = *(const uint4*)gk;
      *(uint4*)&Vt[buf][row][seg * 8] = *(const uint4*)gv;
#endif
    }
  };

  // ---- Q A-fragments (f32 -> f16); 1/sqrt(D) * log2(e) folded in ----
  H16 a0, a1;
  {
    const float* qrow = Q + base + (size_t)(qbase + n) * DHEAD;
    const float qs = 0.125f * LOG2E;
#pragma unroll
    for (int g = 0; g < 2; ++g) {
      float4 qa = *(const float4*)(qrow + g * 16 + a_b0);
      float4 qb = *(const float4*)(qrow + g * 16 + a_b0 + 4);
      float4 qc = *(const float4*)(qrow + 32 + g * 16 + a_b0);
      float4 qd = *(const float4*)(qrow + 32 + g * 16 + a_b0 + 4);
      a0.h[g*8+0] = (_Float16)(qa.x*qs); a0.h[g*8+1] = (_Float16)(qa.y*qs);
      a0.h[g*8+2] = (_Float16)(qa.z*qs); a0.h[g*8+3] = (_Float16)(qa.w*qs);
      a0.h[g*8+4] = (_Float16)(qb.x*qs); a0.h[g*8+5] = (_Float16)(qb.y*qs);
      a0.h[g*8+6] = (_Float16)(qb.z*qs); a0.h[g*8+7] = (_Float16)(qb.w*qs);
      a1.h[g*8+0] = (_Float16)(qc.x*qs); a1.h[g*8+1] = (_Float16)(qc.y*qs);
      a1.h[g*8+2] = (_Float16)(qc.z*qs); a1.h[g*8+3] = (_Float16)(qc.w*qs);
      a1.h[g*8+4] = (_Float16)(qd.x*qs); a1.h[g*8+5] = (_Float16)(qd.y*qs);
      a1.h[g*8+6] = (_Float16)(qd.z*qs); a1.h[g*8+7] = (_Float16)(qd.w*qs);
    }
  }

  v8f O[4] = {v8f{}, v8f{}, v8f{}, v8f{}};
  float m_i[8], l_i[8];
#pragma unroll
  for (int r = 0; r < 8; ++r) { m_i[r] = -1e30f; l_i[r] = 0.0f; }

  issueStage(0);   // prologue: start DMA for chunk 0

  for (int it = 0; it < NCHUNK; ++it) {
    const int kc0 = it * KCHUNK, buf = it & 1;

    // publish chunk `it`: wait own async copies, then barrier
#if USE_ASYNC_LDS
#if __has_builtin(__builtin_amdgcn_s_wait_asynccnt)
    __builtin_amdgcn_s_wait_asynccnt(0);
#else
    asm volatile("s_wait_asynccnt 0x0" ::: "memory");
#endif
#endif
    __syncthreads();

    if (it + 1 < NCHUNK) issueStage(it + 1);   // overlap next chunk's DMA

    const _Float16 (*KtC)[LDSTR] = Kt[buf];
    const _Float16 (*VtC)[LDSTR] = Vt[buf];

    // ---- scores for all 64 keys: four 16x16 tiles, 8 WMMA, pipelined B feeds ----
    auto loadKtile = [&](int t, H16* b) {
#pragma unroll
      for (int j = 0; j < 8; ++j) {
        b[0].u[j] = *(const unsigned*)&KtC[t * 16 + n][     (j >> 2) * 16 + a_b0 + (j & 3) * 2];
        b[1].u[j] = *(const unsigned*)&KtC[t * 16 + n][32 + (j >> 2) * 16 + a_b0 + (j & 3) * 2];
      }
    };
    H16 bk[2][2];
    v8f c[4];
    loadKtile(0, bk[0]);
#pragma unroll
    for (int t = 0; t < 4; ++t) {
      if (t < 3) loadKtile(t + 1, bk[(t + 1) & 1]);
      v8f acc = {};
      acc = wmma_f16(a0.v, bk[t & 1][0].v, acc);
      acc = wmma_f16(a1.v, bk[t & 1][1].v, acc);
      c[t] = acc;
    }

    // mask add (log2 domain)
    float mv[4];
#pragma unroll
    for (int t = 0; t < 4; ++t) mv[t] = M[bidx * S_LEN + kc0 + t * 16 + n] * LOG2E;
#pragma unroll
    for (int t = 0; t < 4; ++t)
#pragma unroll
      for (int r = 0; r < 8; ++r) c[t][r] += mv[t];

    // ---- one online-softmax update for the whole 64-key chunk (log2 domain) ----
#pragma unroll
    for (int r = 0; r < 8; ++r) {
      float rm = fmaxf(fmaxf(c[0][r], c[1][r]), fmaxf(c[2][r], c[3][r]));
      rm = bfly_max<1>(rm); rm = bfly_max<2>(rm);
      rm = bfly_max<4>(rm); rm = bfly_max<8>(rm);
      const float mn = fmaxf(m_i[r], rm);
      const float al = __builtin_amdgcn_exp2f(m_i[r] - mn);
      float ls = 0.0f;
#pragma unroll
      for (int t = 0; t < 4; ++t) {
        const float pv = __builtin_amdgcn_exp2f(c[t][r] - mn);
        ls += pv;
        Pb[wave][r + a_b0][t * 16 + n] = (_Float16)pv;
      }
      float rs = ls;
      rs = bfly_add<1>(rs); rs = bfly_add<2>(rs);
      rs = bfly_add<4>(rs); rs = bfly_add<8>(rs);
      l_i[r] = l_i[r] * al + rs;
      m_i[r] = mn;
      O[0][r] *= al; O[1][r] *= al; O[2][r] *= al; O[3][r] *= al;
    }
    // No barrier: Pb is wave-private; CDNA5 DS ops of a wave complete in order.

    // ---- O += P x V over two 32-key slabs, 8 WMMA, pipelined vb feeds ----
    auto loadVtile = [&](int pair, int jt, H16& b) {
#pragma unroll
      for (int j = 0; j < 8; ++j)
        b.u[j] = *(const unsigned*)&VtC[jt * 16 + n][pair * 32 + (j >> 2) * 16 + a_b0 + (j & 3) * 2];
    };
    H16 pa[2];
#pragma unroll
    for (int pr = 0; pr < 2; ++pr)
#pragma unroll
      for (int j = 0; j < 8; ++j) {
        const int kk = pr * 32 + (j >> 2) * 16 + a_b0 + (j & 3) * 2;
        pa[pr].u[j] = *(const unsigned*)&Pb[wave][n][kk];
      }
    H16 vb[2];
    loadVtile(0, 0, vb[0]);
#pragma unroll
    for (int u = 0; u < 8; ++u) {
      const int pair = u >> 2, jt = u & 3;
      if (u < 7) loadVtile((u + 1) >> 2, (u + 1) & 3, vb[(u + 1) & 1]);
      O[jt] = wmma_f16(pa[pair].v, vb[u & 1].v, O[jt]);
    }
    // No tail barrier: next iteration's wait+barrier orders buffer reuse.
  }

  // ---- epilogue: O / l, f32 store in C layout ----
#pragma unroll
  for (int r = 0; r < 8; ++r) {
    const float inv = 1.0f / l_i[r];
    const size_t orow = base + (size_t)(qbase + r + a_b0) * DHEAD;
#pragma unroll
    for (int jt = 0; jt < 4; ++jt)
      Out[orow + jt * 16 + n] = O[jt][r] * inv;
  }
}

extern "C" void kernel_launch(void* const* d_in, const int* in_sizes, int n_in,
                              void* d_out, int out_size, void* d_ws, size_t ws_size,
                              hipStream_t stream) {
  const float* q = (const float*)d_in[0];
  const float* k = (const float*)d_in[1];
  const float* v = (const float*)d_in[2];
  const float* m = (const float*)d_in[3];
  float* out = (float*)d_out;

  const size_t nelem = (size_t)32 * S_LEN * DHEAD;          // 4.19M
  _Float16* Kh  = (_Float16*)d_ws;                          // 8.39 MB
  _Float16* Vth = (_Float16*)((char*)d_ws + nelem * sizeof(_Float16));

  cvt_k_kernel<<<1024, 256, 0, stream>>>(k, Kh, (int)(nelem / 4));
  cvt_vt_kernel<<<dim3(S_LEN / 64, 32), 256, 0, stream>>>(v, Vth);

  dim3 grid(S_LEN / QTILE, 32);   // 16 q-blocks x (B*H)=32
  fa_fwd_kernel<<<grid, 256, 0, stream>>>(q, Kh, Vth, m, out);
}
